// Beit_14826227106251
// MI455X (gfx1250) — matrix operand order repaired and checked by reading, not compiled, not run on hardware
//
#include <hip/hip_runtime.h>
#include <math.h>

// ---------------------------------------------------------------------------
// BEiT-Base forward for MI455X (gfx1250, wave32, WMMA).
// All GEMMs: LDS-staged block tiles -> v_wmma_f32_16x16x32_f16 (fp32 accum).
// Block tile 128x128, K-step 32; 8 waves as 4(M)x2(N), each wave owns a
// 2x4 grid of 16x16 tiles (8 WMMA per K-step per wave).
// Fills use float4 (b128) fast paths; OOB/unaligned fall back to a
// branchless zero-page address-select scalar path.
// ---------------------------------------------------------------------------

typedef __attribute__((ext_vector_type(16))) _Float16 v16h;
typedef __attribute__((ext_vector_type(8)))  _Float16 v8h;
typedef __attribute__((ext_vector_type(8)))  float    v8f;

namespace {
constexpr int kB   = 4;
constexpr int kCIN = 3;
constexpr int kIMG = 384;
constexpr int kP   = 16;
constexpr int kG   = kIMG / kP;        // 24
constexpr int kNP  = kG * kG;          // 576
constexpr int kN   = kNP + 1;          // 577 tokens
constexpr int kNS  = 580;              // attn row stride (pad to mult of 4)
constexpr int kD   = 768;
constexpr int kH   = 12;
constexpr int kL   = 12;
constexpr int kMLP = 3072;
constexpr int kNC  = 1000;
constexpr int kHD  = kD / kH;          // 64
constexpr int kPATCH = kCIN * kP * kP; // 768
constexpr int kNUMREL = (2*kG-1)*(2*kG-1) + 3; // 2212
constexpr int kROWS = kB * kN;         // 2308
constexpr float kSCALE = 0.125f;       // 64^-0.5
constexpr float kEPS = 1e-6f;

// LDS tile geometry
constexpr int BM = 128, BN = 128, BK = 32;
constexpr int LDA_S = BK + 8;          // 40 halves: pad vs bank conflicts
constexpr int SA_ELEMS = BM * LDA_S;   // 5120 halves
constexpr int SB_ELEMS = BN * LDA_S;   // 5120 halves
constexpr int MT = 2;                  // M tiles per wave (2 x 16 = 32 rows)
constexpr int NT = 4;                  // N tiles per wave (4 x 16 = 64 cols)
}

__device__ __forceinline__ v16h cat8(v8h a, v8h b) {
  return __builtin_shufflevector(a, b, 0,1,2,3,4,5,6,7,8,9,10,11,12,13,14,15);
}

__device__ __forceinline__ v8h cvt8(float4 a, float4 b) {
  v8h r;
  r[0] = (_Float16)a.x; r[1] = (_Float16)a.y;
  r[2] = (_Float16)a.z; r[3] = (_Float16)a.w;
  r[4] = (_Float16)b.x; r[5] = (_Float16)b.y;
  r[6] = (_Float16)b.z; r[7] = (_Float16)b.w;
  return r;
}

__device__ __forceinline__ v8f wmma_acc(v16h a, v16h b, v8f c) {
  return __builtin_amdgcn_wmma_f32_16x16x32_f16(
      false, a, false, b, (short)0, c, false, false);
}

// ---------------------------------------------------------------------------
// Core: one block computes a BM x BN tile of C = A * B.
// A: M x K row-major, stride lda.  B element (k,n) at Bp[k*ldbK + n*ldbN].
// cacc[mt][nt] is the wave's 2x4 grid of 16x16 fp32 accumulators.
// ---------------------------------------------------------------------------
__device__ __forceinline__ void block_gemm_core(
    const float* __restrict__ A, long lda,
    const float* __restrict__ Bp, long ldbK, long ldbN,
    int M, int Ncols, int K, int tileM0, int tileN0,
    const float* __restrict__ zeropad,
    _Float16* __restrict__ sA, _Float16* __restrict__ sB,
    v8f cacc[MT][NT]) {
  const int t   = threadIdx.x;
  const int l   = t & 31;
  const int wv  = t >> 5;
  const int wM  = wv >> 1;             // 0..3
  const int wN  = wv & 1;              // 0..1
  // fill-phase coordinates (A rows / B contiguous-K layout): 16 elems/thread
  const int ar  = t >> 1;              // A row 0..127
  const int ak  = (t & 1) * 16;        // A k-chunk
  const int fn  = t >> 1;              // B col 0..127
  const int fbk = (t & 1) * 16;        // B k-chunk
  // fill-phase coordinates (B strided-K layout: 16 consecutive n, one k)
  const int skk = t >> 3;              // k 0..31
  const int sn0 = (t & 7) * 16;        // n chunk
  const int grow = tileM0 + ar;
  const int gcol = tileN0 + fn;
  const bool rowOk = grow < M;
  const bool colOk = gcol < Ncols;

  const bool alignA  = ((((size_t)A)  & 15) == 0) && ((lda  & 3) == 0);
  const bool contigB = (ldbK == 1);
  const bool alignB  = contigB && ((((size_t)Bp) & 15) == 0) && ((ldbN & 3) == 0);
  const bool alignBs = (!contigB) && ((((size_t)Bp) & 15) == 0) &&
                       ((ldbK & 3) == 0) && (ldbN == 1) && ((tileN0 & 3) == 0);

  for (int k0 = 0; k0 < K; k0 += BK) {
    // ---- stage A tile (row-major, f32 -> f16) ----
    {
      const float* base = A + (size_t)grow * lda + (k0 + ak);
      if (alignA && rowOk && (k0 + ak + 16 <= K)) {
        const float4* p = (const float4*)base;
        *(v8h*)&sA[ar * LDA_S + ak]     = cvt8(p[0], p[1]);
        *(v8h*)&sA[ar * LDA_S + ak + 8] = cvt8(p[2], p[3]);
      } else {
        v8h pk0, pk1;
#pragma unroll
        for (int j = 0; j < 8; ++j) {
          const bool ok0 = rowOk && ((k0 + ak + j) < K);
          const float* p0 = ok0 ? (base + j) : zeropad;
          pk0[j] = (_Float16)(*p0);
          const bool ok1 = rowOk && ((k0 + ak + 8 + j) < K);
          const float* p1 = ok1 ? (base + 8 + j) : zeropad;
          pk1[j] = (_Float16)(*p1);
        }
        *(v8h*)&sA[ar * LDA_S + ak]     = pk0;
        *(v8h*)&sA[ar * LDA_S + ak + 8] = pk1;
      }
    }
    // ---- stage B tile transposed to [n][k] (f32 -> f16) ----
    if (contigB) {
      const float* base = Bp + (size_t)gcol * ldbN + (k0 + fbk);
      if (alignB && colOk && (k0 + fbk + 16 <= K)) {
        const float4* p = (const float4*)base;
        *(v8h*)&sB[fn * LDA_S + fbk]     = cvt8(p[0], p[1]);
        *(v8h*)&sB[fn * LDA_S + fbk + 8] = cvt8(p[2], p[3]);
      } else {
        v8h pk0, pk1;
#pragma unroll
        for (int j = 0; j < 8; ++j) {
          const bool ok0 = colOk && ((k0 + fbk + j) < K);
          const float* p0 = ok0 ? (base + j) : zeropad;
          pk0[j] = (_Float16)(*p0);
          const bool ok1 = colOk && ((k0 + fbk + 8 + j) < K);
          const float* p1 = ok1 ? (base + 8 + j) : zeropad;
          pk1[j] = (_Float16)(*p1);
        }
        *(v8h*)&sB[fn * LDA_S + fbk]     = pk0;
        *(v8h*)&sB[fn * LDA_S + fbk + 8] = pk1;
      }
    } else {
      const int gk = k0 + skk;
      const int gn = tileN0 + sn0;
      const float* base = Bp + (size_t)gk * ldbK + (size_t)gn * ldbN;
      _Float16* d = &sB[sn0 * LDA_S + skk];
      if (alignBs && (gk < K) && (gn + 16 <= Ncols)) {
        const float4* p = (const float4*)base;
        const float4 q0 = p[0], q1 = p[1], q2 = p[2], q3 = p[3];
        d[ 0*LDA_S] = (_Float16)q0.x; d[ 1*LDA_S] = (_Float16)q0.y;
        d[ 2*LDA_S] = (_Float16)q0.z; d[ 3*LDA_S] = (_Float16)q0.w;
        d[ 4*LDA_S] = (_Float16)q1.x; d[ 5*LDA_S] = (_Float16)q1.y;
        d[ 6*LDA_S] = (_Float16)q1.z; d[ 7*LDA_S] = (_Float16)q1.w;
        d[ 8*LDA_S] = (_Float16)q2.x; d[ 9*LDA_S] = (_Float16)q2.y;
        d[10*LDA_S] = (_Float16)q2.z; d[11*LDA_S] = (_Float16)q2.w;
        d[12*LDA_S] = (_Float16)q3.x; d[13*LDA_S] = (_Float16)q3.y;
        d[14*LDA_S] = (_Float16)q3.z; d[15*LDA_S] = (_Float16)q3.w;
      } else {
#pragma unroll
        for (int i = 0; i < 16; ++i) {
          const bool ok = (gk < K) && ((gn + i) < Ncols);
          const float* p = ok ? (base + (size_t)i * ldbN) : zeropad;
          d[i * LDA_S] = (_Float16)(*p);
        }
      }
    }
    // speculative prefetch of next K-tile (global_prefetch_b8)
    if (k0 + BK < K) {
      __builtin_prefetch(A + (size_t)grow * lda + (k0 + BK + ak), 0, 3);
      if (contigB)
        __builtin_prefetch(Bp + (size_t)gcol * ldbN + (k0 + BK + fbk), 0, 3);
      else
        __builtin_prefetch(Bp + (size_t)(k0 + BK + skk) * ldbK
                              + (size_t)(tileN0 + sn0) * ldbN, 0, 3);
    }
    __syncthreads();

    // ---- fragments from LDS + WMMA (2 A frags, 4 B frags, 8 WMMA) ----
    const int kbA = (l >> 4) * 16;
    const int kbB = (l >> 4) * 16;
    v16h af[MT], bf[NT];
#pragma unroll
    for (int mt = 0; mt < MT; ++mt) {
      const int r = wM * 32 + mt * 16 + (l & 15);
      const _Float16* p = sA + r * LDA_S + kbA;
      af[mt] = cat8(*(const v8h*)p, *(const v8h*)(p + 8));
    }
#pragma unroll
    for (int nt = 0; nt < NT; ++nt) {
      const int n = wN * 64 + nt * 16 + (l & 15);
      const _Float16* p = sB + n * LDA_S + kbB;
      bf[nt] = cat8(*(const v8h*)p, *(const v8h*)(p + 8));
    }
#pragma unroll
    for (int mt = 0; mt < MT; ++mt)
#pragma unroll
      for (int nt = 0; nt < NT; ++nt)
        cacc[mt][nt] = wmma_acc(af[mt], bf[nt], cacc[mt][nt]);
    __syncthreads();
  }
}

// ---------------- Generic GEMM: C = A*B (+bias)(+resid)(+GELU) -------------
__global__ __launch_bounds__(256)
void gemm_wmma(const float* __restrict__ A, long lda,
               const float* __restrict__ Bp, long ldbK, long ldbN,
               float* __restrict__ C, long ldc,
               const float* __restrict__ bias,
               const float* __restrict__ resid, long ldr,
               int M, int Ncols, int K, int fuse_gelu,
               const float* __restrict__ zeropad) {
  __shared__ __align__(16) _Float16 sA[SA_ELEMS];
  __shared__ __align__(16) _Float16 sB[SB_ELEMS];
  const int tileM0 = blockIdx.x * BM;
  const int tileN0 = blockIdx.y * BN;
  v8f cacc[MT][NT] = {};
  block_gemm_core(A, lda, Bp, ldbK, ldbN, M, Ncols, K, tileM0, tileN0,
                  zeropad, sA, sB, cacc);

  const int l = threadIdx.x & 31;
  const int wv = threadIdx.x >> 5, wM = wv >> 1, wN = wv & 1;
#pragma unroll
  for (int mt = 0; mt < MT; ++mt)
#pragma unroll
    for (int nt = 0; nt < NT; ++nt) {
      const int col   = tileN0 + wN * 64 + nt * 16 + (l & 15);
      const int rbase = tileM0 + wM * 32 + mt * 16 + 8 * (l >> 4);
#pragma unroll
      for (int r = 0; r < 8; ++r) {
        const int row = rbase + r;
        if (row < M && col < Ncols) {
          float v = cacc[mt][nt][r];
          if (bias)  v += bias[col];
          if (resid) v += resid[(size_t)row * ldr + col];
          if (fuse_gelu) v = 0.5f * v * (1.f + erff(v * 0.70710678118654752f));
          C[(size_t)row * ldc + col] = v;
        }
      }
    }
}

// ---------------- Attention scores: attn = SCALE*Q.K^T + relbias -----------
__global__ __launch_bounds__(256)
void attn_score_kernel(const float* __restrict__ qkv, float* __restrict__ attn,
                       const float* __restrict__ rel_t,
                       const int* __restrict__ rel_idx,
                       const float* __restrict__ zeropad) {
  __shared__ __align__(16) _Float16 sA[SA_ELEMS];
  __shared__ __align__(16) _Float16 sB[SB_ELEMS];
  const int z = blockIdx.z, b = z / kH, h = z % kH;
  const int tileM0 = blockIdx.x * BM;   // query tile
  const int tileN0 = blockIdx.y * BN;   // key tile
  const float* qb = qkv + (size_t)b * kN * 3 * kD + (size_t)h * kHD;
  const float* kb = qb + kD;
  v8f cacc[MT][NT] = {};
  block_gemm_core(qb, 3 * kD, kb, 1, 3 * kD, kN, kN, kHD, tileM0, tileN0,
                  zeropad, sA, sB, cacc);

  float* ap = attn + (size_t)z * kN * kNS;
  const int l = threadIdx.x & 31;
  const int wv = threadIdx.x >> 5, wM = wv >> 1, wN = wv & 1;
#pragma unroll
  for (int mt = 0; mt < MT; ++mt)
#pragma unroll
    for (int nt = 0; nt < NT; ++nt) {
      const int col   = tileN0 + wN * 64 + nt * 16 + (l & 15);
      const int rbase = tileM0 + wM * 32 + mt * 16 + 8 * (l >> 4);
#pragma unroll
      for (int r = 0; r < 8; ++r) {
        const int row = rbase + r;
        if (row < kN && col < kN) {
          const int ri = rel_idx[(size_t)row * kN + col];
          ap[(size_t)row * kNS + col] =
              cacc[mt][nt][r] * kSCALE + rel_t[(size_t)ri * kH + h];
        }
      }
    }
}

// ---------------- attn @ V  -> o (B, N, D) ---------------------------------
__global__ __launch_bounds__(256)
void attn_v_kernel(const float* __restrict__ attn, const float* __restrict__ qkv,
                   float* __restrict__ o, const float* __restrict__ zeropad) {
  __shared__ __align__(16) _Float16 sA[SA_ELEMS];
  __shared__ __align__(16) _Float16 sB[SB_ELEMS];
  const int z = blockIdx.z, b = z / kH, h = z % kH;
  const int tileM0 = blockIdx.x * BM;
  const float* ap = attn + (size_t)z * kN * kNS;
  const float* vb = qkv + (size_t)b * kN * 3 * kD + 2 * kD + (size_t)h * kHD;
  v8f cacc[MT][NT] = {};
  block_gemm_core(ap, kNS, vb, 3 * kD, 1, kN, kHD, kN, tileM0, 0,
                  zeropad, sA, sB, cacc);

  float* ob = o + (size_t)b * kN * kD + (size_t)h * kHD;
  const int l = threadIdx.x & 31;
  const int wv = threadIdx.x >> 5, wM = wv >> 1, wN = wv & 1;
#pragma unroll
  for (int mt = 0; mt < MT; ++mt)
#pragma unroll
    for (int nt = 0; nt < NT; ++nt) {
      const int col   = wN * 64 + nt * 16 + (l & 15);
      const int rbase = tileM0 + wM * 32 + mt * 16 + 8 * (l >> 4);
#pragma unroll
      for (int r = 0; r < 8; ++r) {
        const int row = rbase + r;
        if (row < kN && col < kHD)
          ob[(size_t)row * kD + col] = cacc[mt][nt][r];
      }
    }
}

// ---------------- LayerNorm: one block per row ------------------------------
__global__ __launch_bounds__(256)
void ln_kernel(const float* __restrict__ in, float* __restrict__ out,
               const float* __restrict__ w, const float* __restrict__ b, int D) {
  __shared__ float red[256];
  const int row = blockIdx.x;
  const float* xr = in + (size_t)row * D;

  float s = 0.f;
  for (int i = threadIdx.x; i < D; i += 256) s += xr[i];
  red[threadIdx.x] = s; __syncthreads();
  for (int o2 = 128; o2 > 0; o2 >>= 1) {
    if (threadIdx.x < o2) red[threadIdx.x] += red[threadIdx.x + o2];
    __syncthreads();
  }
  const float mean = red[0] / (float)D;
  __syncthreads();

  float vs = 0.f;
  for (int i = threadIdx.x; i < D; i += 256) { float d = xr[i] - mean; vs += d * d; }
  red[threadIdx.x] = vs; __syncthreads();
  for (int o2 = 128; o2 > 0; o2 >>= 1) {
    if (threadIdx.x < o2) red[threadIdx.x] += red[threadIdx.x + o2];
    __syncthreads();
  }
  const float inv = rsqrtf(red[0] / (float)D + kEPS);

  for (int i = threadIdx.x; i < D; i += 256)
    out[(size_t)row * D + i] = (xr[i] - mean) * inv * w[i] + b[i];
}

// ---------------- Softmax: one wave32 per row (stride kNS, len kN) ---------
__global__ __launch_bounds__(256)
void softmax_kernel(float* __restrict__ attn, int rows) {
  const int row = blockIdx.x * 8 + (threadIdx.x >> 5);
  if (row >= rows) return;
  float* p = attn + (size_t)row * kNS;
  const int l = threadIdx.x & 31;

  float mx = -3.4e38f;
  for (int i = l; i < kN; i += 32) mx = fmaxf(mx, p[i]);
#pragma unroll
  for (int o2 = 16; o2 > 0; o2 >>= 1) mx = fmaxf(mx, __shfl_xor(mx, o2, 32));

  float sum = 0.f;
  for (int i = l; i < kN; i += 32) { float e = __expf(p[i] - mx); p[i] = e; sum += e; }
#pragma unroll
  for (int o2 = 16; o2 > 0; o2 >>= 1) sum += __shfl_xor(sum, o2, 32);

  const float inv = 1.f / sum;
  for (int i = l; i < kN; i += 32) p[i] *= inv;
}

// ---------------- Elementwise helpers --------------------------------------
__global__ void zero_kernel(float* __restrict__ p, int n) {
  const int i = blockIdx.x * blockDim.x + threadIdx.x;
  if (i < n) p[i] = 0.f;
}

__global__ void gather_patches_kernel(const float* __restrict__ x,
                                      float* __restrict__ patches) {
  const size_t idx = (size_t)blockIdx.x * blockDim.x + threadIdx.x;
  const size_t tot = (size_t)kB * kNP * kPATCH;
  if (idx >= tot) return;
  const int col = (int)(idx % kPATCH);
  const size_t t = idx / kPATCH;
  const int np = (int)(t % kNP);
  const int b  = (int)(t / kNP);
  const int c  = col / (kP * kP);
  const int pr = (col / kP) % kP;
  const int pc = col % kP;
  const int gy = np / kG, gx = np % kG;
  patches[idx] = x[(((size_t)b * kCIN + c) * kIMG + gy * kP + pr) * kIMG
                   + gx * kP + pc];
}

__global__ void assemble_h_kernel(const float* __restrict__ tok,
                                  const float* __restrict__ cls,
                                  const float* __restrict__ pos,
                                  float* __restrict__ h) {
  const size_t idx = (size_t)blockIdx.x * blockDim.x + threadIdx.x;
  const size_t tot = (size_t)kROWS * kD;
  if (idx >= tot) return;
  const int d = (int)(idx % kD);
  const size_t t = idx / kD;
  const int r = (int)(t % kN);
  const int b = (int)(t / kN);
  const float v = (r == 0) ? cls[d]
                           : tok[((size_t)b * kNP + (r - 1)) * kD + d];
  h[idx] = v + pos[(size_t)r * kD + d];
}

__global__ void build_qkv_bias_kernel(const float* __restrict__ qb,
                                      const float* __restrict__ vb,
                                      float* __restrict__ qkvb) {
  const int idx = blockIdx.x * blockDim.x + threadIdx.x;
  if (idx >= kL * 3 * kD) return;
  const int c = idx % (3 * kD);
  const int l = idx / (3 * kD);
  float v = 0.f;
  if (c < kD)            v = qb[(size_t)l * kD + c];
  else if (c >= 2 * kD)  v = vb[(size_t)l * kD + (c - 2 * kD)];
  qkvb[idx] = v;
}

__global__ void pool_kernel(const float* __restrict__ h, float* __restrict__ pooled) {
  const int idx = blockIdx.x * blockDim.x + threadIdx.x;
  if (idx >= kB * kD) return;
  const int b = idx / kD, d = idx % kD;
  float s = 0.f;
  for (int n = 1; n < kN; ++n) s += h[((size_t)b * kN + n) * kD + d];
  pooled[idx] = s / (float)kNP;
}

// ---------------------------------------------------------------------------
extern "C" void kernel_launch(void* const* d_in, const int* in_sizes, int n_in,
                              void* d_out, int out_size, void* d_ws, size_t ws_size,
                              hipStream_t stream) {
  (void)in_sizes; (void)n_in; (void)out_size; (void)ws_size;
  const float* x         = (const float*)d_in[0];
  const float* patch_w   = (const float*)d_in[1];
  const float* patch_b   = (const float*)d_in[2];
  const float* cls_token = (const float*)d_in[3];
  const float* pos_embed = (const float*)d_in[4];
  const float* ln1_w     = (const float*)d_in[5];
  const float* ln1_b     = (const float*)d_in[6];
  const float* qkv_w     = (const float*)d_in[7];
  const float* q_bias    = (const float*)d_in[8];
  const float* v_bias    = (const float*)d_in[9];
  const float* proj_w    = (const float*)d_in[10];
  const float* proj_b    = (const float*)d_in[11];
  const float* ln2_w     = (const float*)d_in[12];
  const float* ln2_b     = (const float*)d_in[13];
  const float* fc1_w     = (const float*)d_in[14];
  const float* fc1_b     = (const float*)d_in[15];
  const float* fc2_w     = (const float*)d_in[16];
  const float* fc2_b     = (const float*)d_in[17];
  const float* rel_table = (const float*)d_in[18];
  const float* fc_norm_w = (const float*)d_in[19];
  const float* fc_norm_b = (const float*)d_in[20];
  const float* head_w    = (const float*)d_in[21];
  const float* head_b    = (const float*)d_in[22];
  const int*   rel_index = (const int*)d_in[23];
  float* out = (float*)d_out;

  // ---- workspace carve-out (fp32) ----
  float* ws = (float*)d_ws;
  size_t off = 0;
  auto carve = [&](size_t nelem) { float* p = ws + off; off += nelem; return p; };
  float* patches = carve((size_t)kB * kNP * kPATCH);
  float* tok     = carve((size_t)kB * kNP * kD);
  float* h       = carve((size_t)kROWS * kD);
  float* y       = carve((size_t)kROWS * kD);
  float* qkv     = carve((size_t)kROWS * 3 * kD);
  float* attnb   = carve((size_t)kB * kH * kN * kNS);
  float* o       = carve((size_t)kROWS * kD);
  float* mlp     = carve((size_t)kROWS * kMLP);
  float* qkvb    = carve((size_t)kL * 3 * kD);
  float* pooled  = carve((size_t)kB * kD);
  float* pooledn = carve((size_t)kB * kD);
  float* zeropad = carve(64);

  zero_kernel<<<1, 64, 0, stream>>>(zeropad, 64);

  // ---- patch embed ----
  {
    const size_t tot = (size_t)kB * kNP * kPATCH;
    gather_patches_kernel<<<(unsigned)((tot + 255) / 256), 256, 0, stream>>>(x, patches);
  }
  {
    dim3 g((kB * kNP + BM - 1) / BM, (kD + BN - 1) / BN);
    gemm_wmma<<<g, 256, 0, stream>>>(patches, kPATCH, patch_w, 1, kPATCH,
                                     tok, kD, patch_b, nullptr, 0,
                                     kB * kNP, kD, kPATCH, 0, zeropad);
  }
  {
    const size_t tot = (size_t)kROWS * kD;
    assemble_h_kernel<<<(unsigned)((tot + 255) / 256), 256, 0, stream>>>(
        tok, cls_token, pos_embed, h);
  }
  build_qkv_bias_kernel<<<(kL * 3 * kD + 255) / 256, 256, 0, stream>>>(
      q_bias, v_bias, qkvb);

  // ---- transformer layers ----
  for (int l = 0; l < kL; ++l) {
    ln_kernel<<<kROWS, 256, 0, stream>>>(h, y, ln1_w + (size_t)l * kD,
                                         ln1_b + (size_t)l * kD, kD);
    {
      dim3 g((kROWS + BM - 1) / BM, (3 * kD + BN - 1) / BN);
      gemm_wmma<<<g, 256, 0, stream>>>(y, kD, qkv_w + (size_t)l * 3 * kD * kD, 1, kD,
                                       qkv, 3 * kD, qkvb + (size_t)l * 3 * kD,
                                       nullptr, 0, kROWS, 3 * kD, kD, 0, zeropad);
    }
    {
      dim3 g((kN + BM - 1) / BM, (kN + BN - 1) / BN, kB * kH);
      attn_score_kernel<<<g, 256, 0, stream>>>(
          qkv, attnb, rel_table + (size_t)l * kNUMREL * kH, rel_index, zeropad);
    }
    {
      const int rows = kB * kH * kN;
      softmax_kernel<<<(rows + 7) / 8, 256, 0, stream>>>(attnb, rows);
    }
    {
      dim3 g((kN + BM - 1) / BM, 1, kB * kH);
      attn_v_kernel<<<g, 256, 0, stream>>>(attnb, qkv, o, zeropad);
    }
    {
      dim3 g((kROWS + BM - 1) / BM, (kD + BN - 1) / BN);
      gemm_wmma<<<g, 256, 0, stream>>>(o, kD, proj_w + (size_t)l * kD * kD, 1, kD,
                                       h, kD, proj_b + (size_t)l * kD, h, kD,
                                       kROWS, kD, kD, 0, zeropad);
    }
    ln_kernel<<<kROWS, 256, 0, stream>>>(h, y, ln2_w + (size_t)l * kD,
                                         ln2_b + (size_t)l * kD, kD);
    {
      dim3 g((kROWS + BM - 1) / BM, (kMLP + BN - 1) / BN);
      gemm_wmma<<<g, 256, 0, stream>>>(y, kD, fc1_w + (size_t)l * kMLP * kD, 1, kD,
                                       mlp, kMLP, fc1_b + (size_t)l * kMLP,
                                       nullptr, 0, kROWS, kMLP, kD, 1, zeropad);
    }
    {
      dim3 g((kROWS + BM - 1) / BM, (kD + BN - 1) / BN);
      gemm_wmma<<<g, 256, 0, stream>>>(mlp, kMLP, fc2_w + (size_t)l * kD * kMLP, 1, kMLP,
                                       h, kD, fc2_b + (size_t)l * kD, h, kD,
                                       kROWS, kD, kMLP, 0, zeropad);
    }
  }

  // ---- head: mean-pool (drop cls) -> LN -> linear ----
  pool_kernel<<<(kB * kD + 255) / 256, 256, 0, stream>>>(h, pooled);
  ln_kernel<<<kB, 256, 0, stream>>>(pooled, pooledn, fc_norm_w, fc_norm_b, kD);
  {
    dim3 g(1, (kNC + BN - 1) / BN);
    gemm_wmma<<<g, 256, 0, stream>>>(pooledn, kD, head_w, 1, kD,
                                     out, kNC, head_b, nullptr, 0,
                                     kB, kNC, kD, 0, zeropad);
  }
}
